// LSTMEncoder_12979391168879
// MI455X (gfx1250) — matrix operand order repaired and checked by reading, not compiled, not run on hardware
//
#include <hip/hip_runtime.h>
#include <hip/hip_bf16.h>

// ---------------------------------------------------------------------------
// Problem constants (from reference)
// ---------------------------------------------------------------------------
#define NSPEC   16
#define IDIM    256
#define HDIM    512
#define ODIM    1024      // MSG_DIM * VOCAB
#define AROWS   512
#define BCOLS   32
#define MROWS   (AROWS * BCOLS)   // 16384 flattened rows

// ---------------------------------------------------------------------------
// WMMA fragment types (CDNA5 gfx1250, wave32)
// ---------------------------------------------------------------------------
typedef __attribute__((ext_vector_type(16))) __bf16     v16bf;
typedef __attribute__((ext_vector_type(8)))  float      v8f;
typedef __attribute__((ext_vector_type(8)))  unsigned   v8u;
typedef __attribute__((ext_vector_type(4)))  unsigned   v4u;
typedef __attribute__((ext_vector_type(4)))  float      v4f;
typedef __attribute__((ext_vector_type(4)))  unsigned short v4us;

union AFrag { v16bf bf; v4u u[2]; };   // 16x32 bf16 A: two 16B K-runs per lane
union BFrag { v16bf bf; v8u  u;   };   // 32x16 bf16 B: one 32B K-run per lane

__device__ __forceinline__ unsigned short f2bf(float f) {
    unsigned u = __float_as_uint(f);
    unsigned r = u + 0x7FFFu + ((u >> 16) & 1u);   // round-to-nearest-even
    return (unsigned short)(r >> 16);
}
__device__ __forceinline__ float fsigmoid(float x) {
    return 1.0f / (1.0f + __expf(-x));
}
__device__ __forceinline__ float ftanh(float x) {
    // tanh(x) = 1 - 2/(exp(2x)+1); saturates correctly for |x| large
    return 1.0f - 2.0f / (__expf(2.0f * x) + 1.0f);
}
__device__ __forceinline__ v8f wmma_bf16(const AFrag& a, const BFrag& b, v8f c) {
    return __builtin_amdgcn_wmma_f32_16x16x32_bf16(
        false, a.bf, false, b.bf, (short)0, c, false, false);
}
__device__ __forceinline__ void load_a(AFrag& af, const unsigned short* p) {
    af.u[0] = *(const v4u*)(p);
    af.u[1] = *(const v4u*)(p + 8);
}

// ---------------------------------------------------------------------------
// f32 -> bf16 conversion, 4-wide (all array sizes are multiples of 4)
// ---------------------------------------------------------------------------
__global__ void k_cvt4(const v4f* __restrict__ in, v4us* __restrict__ out, int n4) {
    for (int i = blockIdx.x * blockDim.x + threadIdx.x; i < n4; i += gridDim.x * blockDim.x) {
        v4f f = in[i];
        v4us o;
        o.x = f2bf(f.x); o.y = f2bf(f.y); o.z = f2bf(f.z); o.w = f2bf(f.w);
        out[i] = o;
    }
}

// ---------------------------------------------------------------------------
// Kernel 1: x_proj[a,b,h] = relu( x[a] @ w_in[s[a]]^T + b_in[s[a]] ) -> bf16
// Per block: one 'a' (M=32, N=512, K=256). 8 waves: wave w -> m-tile (w>>2),
// n-range (w&3)*128 (8 tiles of 16).
// Pipeline per k-step: load 8 B frags (16 b128, in-order drain under the
// WMMA sequence) ; A fragment double-buffered across k-steps.
// ---------------------------------------------------------------------------
__global__ void __launch_bounds__(256) k_proj(
    const unsigned short* __restrict__ xbf,     // (16384,256) bf16
    const unsigned short* __restrict__ winbf,   // (16,512,256) bf16
    const float* __restrict__ b_in,             // (16,512)
    const int*  __restrict__ species,           // (512,)
    unsigned short* __restrict__ xprojbf)       // (16384,512) bf16
{
    const int a    = blockIdx.x;
    const int lane = threadIdx.x & 31;
    const int wave = threadIdx.x >> 5;
    const int half = lane >> 4;
    const int lm   = lane & 15;
    const int s    = species[a];
    const int mt   = wave >> 2;           // 0..1
    const int n0   = (wave & 3) * 128;

    const unsigned short* Arow = xbf + (size_t)(a * 32 + mt * 16 + lm) * IDIM + 16 * half;
    const unsigned short* W    = winbf + (size_t)s * HDIM * IDIM +
                                 (size_t)(n0 + lm) * IDIM + 16 * half;

    v8f acc[8] = {};
    AFrag af[2];
    load_a(af[0], Arow);
    for (int kt = 0; kt < IDIM / 32; ++kt) {
        const int k0  = kt * 32;
        const int cur = kt & 1;
        BFrag bfr[8];
        #pragma unroll
        for (int t = 0; t < 8; ++t)
            bfr[t].u = *(const v8u*)(W + (size_t)(t * 16) * IDIM + k0);
        if (kt + 1 < IDIM / 32)
            load_a(af[cur ^ 1], Arow + k0 + 32);
        #pragma unroll
        for (int t = 0; t < 8; ++t)
            acc[t] = wmma_bf16(af[cur], bfr[t], acc[t]);
    }
    const float* brow = b_in + s * HDIM;
    #pragma unroll
    for (int t = 0; t < 8; ++t) {
        const int n = n0 + t * 16 + lm;
        const float bias = brow[n];
        unsigned short* orow = xprojbf + (size_t)(a * 32 + mt * 16) * HDIM + n;
        #pragma unroll
        for (int v = 0; v < 8; ++v) {
            const int m = v + 8 * half;
            float val = acc[t][v] + bias;
            val = fmaxf(val, 0.0f);
            orow[(size_t)m * HDIM] = f2bf(val);
        }
    }
}

// ---------------------------------------------------------------------------
// Kernel 2: gates = x_proj @ w_ih^T + b_ih + b_hh ; LSTM pointwise (h0=c0=0).
// gf is never used (c0 == 0), so only gate segments {0:i, 2:g, 3:o} computed.
// Tiling for B-reuse: block = 64 rows (4 m-tiles) x 128 h-columns.
// Wave w owns h-columns [blockIdx.y*128 + w*16, +16) in all 3 live segments.
// Pipeline per k-step: load 4 A frags (8 b128, staggered drain under the 12
// WMMAs); the 3 B fragments are double-buffered across k-steps.
// ---------------------------------------------------------------------------
__global__ void __launch_bounds__(256) k_lstm(
    const unsigned short* __restrict__ xprojbf,  // (16384,512) bf16
    const unsigned short* __restrict__ wihbf,    // (2048,512)  bf16
    const float* __restrict__ b_ih,              // (2048,)
    const float* __restrict__ b_hh,              // (2048,)
    float* __restrict__ h_out,                   // (16384,512)
    float* __restrict__ c_out,                   // (16384,512)
    unsigned short* __restrict__ h1bf)           // (16384,512) bf16
{
    const int mbase = blockIdx.x * 64;           // rows mbase .. mbase+64
    const int lane  = threadIdx.x & 31;
    const int wave  = threadIdx.x >> 5;
    const int half  = lane >> 4;
    const int lm    = lane & 15;
    const int hcol  = blockIdx.y * 128 + wave * 16 + lm;   // column within H

    // B rows for the three live gate segments: i(0), g(2), o(3)
    const unsigned short* Bi = wihbf + (size_t)(0 * HDIM + hcol) * HDIM + 16 * half;
    const unsigned short* Bg = wihbf + (size_t)(2 * HDIM + hcol) * HDIM + 16 * half;
    const unsigned short* Bo = wihbf + (size_t)(3 * HDIM + hcol) * HDIM + 16 * half;
    const unsigned short* Abase = xprojbf + (size_t)(mbase + lm) * HDIM + 16 * half;

    v8f acc[4][3] = {};
    BFrag b[2][3];
    b[0][0].u = *(const v8u*)(Bi);
    b[0][1].u = *(const v8u*)(Bg);
    b[0][2].u = *(const v8u*)(Bo);
    for (int kt = 0; kt < HDIM / 32; ++kt) {
        const int k0  = kt * 32;
        const int cur = kt & 1;
        AFrag af[4];
        #pragma unroll
        for (int mt = 0; mt < 4; ++mt)
            load_a(af[mt], Abase + (size_t)(mt * 16) * HDIM + k0);
        if (kt + 1 < HDIM / 32) {
            b[cur ^ 1][0].u = *(const v8u*)(Bi + k0 + 32);
            b[cur ^ 1][1].u = *(const v8u*)(Bg + k0 + 32);
            b[cur ^ 1][2].u = *(const v8u*)(Bo + k0 + 32);
        }
        #pragma unroll
        for (int mt = 0; mt < 4; ++mt) {
            acc[mt][0] = wmma_bf16(af[mt], b[cur][0], acc[mt][0]);
            acc[mt][1] = wmma_bf16(af[mt], b[cur][1], acc[mt][1]);
            acc[mt][2] = wmma_bf16(af[mt], b[cur][2], acc[mt][2]);
        }
    }
    const float bi = b_ih[0 * HDIM + hcol] + b_hh[0 * HDIM + hcol];
    const float bg = b_ih[2 * HDIM + hcol] + b_hh[2 * HDIM + hcol];
    const float bo = b_ih[3 * HDIM + hcol] + b_hh[3 * HDIM + hcol];
    #pragma unroll
    for (int mt = 0; mt < 4; ++mt) {
        #pragma unroll
        for (int v = 0; v < 8; ++v) {
            const int m = mbase + mt * 16 + v + 8 * half;
            const float iv = fsigmoid(acc[mt][0][v] + bi);
            const float gv = ftanh   (acc[mt][1][v] + bg);
            const float ov = fsigmoid(acc[mt][2][v] + bo);
            const float c1 = iv * gv;                    // f*c0 == 0
            const float h1 = ov * ftanh(c1);
            c_out[(size_t)m * HDIM + hcol] = c1;
            h_out[(size_t)m * HDIM + hcol] = h1;
            h1bf [(size_t)m * HDIM + hcol] = f2bf(h1);
        }
    }
}

// ---------------------------------------------------------------------------
// Kernel 3: logits[a] = h1[a] @ w_out[s[a]]^T + b_out[s[a]]
// Per block: one 'a' (M=32, N=1024, K=512), 512 threads = 16 waves.
// wave w -> m-tile (w>>3), n-range (w&7)*128 (8 tiles of 16).
// Same pipeline idiom as k_proj (8 B frags per k-step, A double-buffered).
// ---------------------------------------------------------------------------
__global__ void __launch_bounds__(512) k_out(
    const unsigned short* __restrict__ h1bf,     // (16384,512) bf16
    const unsigned short* __restrict__ woutbf,   // (16,1024,512) bf16
    const float* __restrict__ b_out,             // (16,1024)
    const int*  __restrict__ species,            // (512,)
    float* __restrict__ logits)                  // (512,32,1024)
{
    const int a    = blockIdx.x;
    const int lane = threadIdx.x & 31;
    const int wave = threadIdx.x >> 5;
    const int half = lane >> 4;
    const int lm   = lane & 15;
    const int s    = species[a];
    const int mt   = wave >> 3;            // 0..1
    const int n0   = (wave & 7) * 128;

    const unsigned short* Arow = h1bf + (size_t)(a * 32 + mt * 16 + lm) * HDIM + 16 * half;
    const unsigned short* W    = woutbf + (size_t)s * ODIM * HDIM +
                                 (size_t)(n0 + lm) * HDIM + 16 * half;

    v8f acc[8] = {};
    AFrag af[2];
    load_a(af[0], Arow);
    for (int kt = 0; kt < HDIM / 32; ++kt) {
        const int k0  = kt * 32;
        const int cur = kt & 1;
        BFrag bfr[8];
        #pragma unroll
        for (int t = 0; t < 8; ++t)
            bfr[t].u = *(const v8u*)(W + (size_t)(t * 16) * HDIM + k0);
        if (kt + 1 < HDIM / 32)
            load_a(af[cur ^ 1], Arow + k0 + 32);
        #pragma unroll
        for (int t = 0; t < 8; ++t)
            acc[t] = wmma_bf16(af[cur], bfr[t], acc[t]);
    }
    const float* brow = b_out + s * ODIM;
    #pragma unroll
    for (int t = 0; t < 8; ++t) {
        const int n = n0 + t * 16 + lm;
        const float bias = brow[n];
        float* orow = logits + (size_t)(a * 32 + mt * 16) * ODIM + n;
        #pragma unroll
        for (int v = 0; v < 8; ++v) {
            const int m = v + 8 * half;
            orow[(size_t)m * ODIM] = acc[t][v] + bias;
        }
    }
}

// ---------------------------------------------------------------------------
// Launch
// ---------------------------------------------------------------------------
extern "C" void kernel_launch(void* const* d_in, const int* in_sizes, int n_in,
                              void* d_out, int out_size, void* d_ws, size_t ws_size,
                              hipStream_t stream) {
    (void)in_sizes; (void)n_in; (void)out_size; (void)ws_size;
    const float* x       = (const float*)d_in[0];   // (512,32,256)
    const int*   species = (const int*)  d_in[1];   // (512,)
    const float* w_in    = (const float*)d_in[2];   // (16,512,256)
    const float* b_in    = (const float*)d_in[3];   // (16,512)
    const float* w_ih    = (const float*)d_in[4];   // (2048,512)
    // d_in[5] = w_hh : unused (h0 == 0)
    const float* b_ih    = (const float*)d_in[6];   // (2048,)
    const float* b_hh    = (const float*)d_in[7];   // (2048,)
    const float* w_out   = (const float*)d_in[8];   // (16,1024,512)
    const float* b_out   = (const float*)d_in[9];   // (16,1024)

    float* out    = (float*)d_out;
    float* logits = out;                                   // 16,777,216 floats
    float* h_out  = out + (size_t)AROWS * BCOLS * ODIM;    //  8,388,608 floats
    float* c_out  = h_out + (size_t)MROWS * HDIM;          //  8,388,608 floats

    // Workspace layout (bf16 buffers), all offsets 256B-aligned
    char* ws = (char*)d_ws;
    unsigned short* xbf     = (unsigned short*)(ws);                     // 16384*256
    unsigned short* winbf   = (unsigned short*)(ws + 8388608);           // 16*512*256
    unsigned short* wihbf   = (unsigned short*)(ws + 12582912);          // 2048*512
    unsigned short* woutbf  = (unsigned short*)(ws + 14680064);          // 16*1024*512
    unsigned short* xprojbf = (unsigned short*)(ws + 31457280);          // 16384*512
    unsigned short* h1bf    = (unsigned short*)(ws + 48234496);          // 16384*512

    const int CT = 256;
    {
        int n4;
        n4 = MROWS * IDIM / 4;
        k_cvt4<<<(n4 + CT - 1) / CT, CT, 0, stream>>>((const v4f*)x, (v4us*)xbf, n4);
        n4 = NSPEC * HDIM * IDIM / 4;
        k_cvt4<<<(n4 + CT - 1) / CT, CT, 0, stream>>>((const v4f*)w_in, (v4us*)winbf, n4);
        n4 = 4 * HDIM * HDIM / 4;
        k_cvt4<<<(n4 + CT - 1) / CT, CT, 0, stream>>>((const v4f*)w_ih, (v4us*)wihbf, n4);
        n4 = NSPEC * ODIM * HDIM / 4;
        k_cvt4<<<(n4 + CT - 1) / CT, CT, 0, stream>>>((const v4f*)w_out, (v4us*)woutbf, n4);
    }

    k_proj<<<AROWS, 256, 0, stream>>>(xbf, winbf, b_in, species, xprojbf);
    k_lstm<<<dim3(MROWS / 64, HDIM / 128), 256, 0, stream>>>(
        xprojbf, wihbf, b_ih, b_hh, h_out, c_out, h1bf);
    k_out <<<AROWS, 512, 0, stream>>>(h1bf, woutbf, b_out, species, logits);
}
